// CaptionLoss_61795989455374
// MI455X (gfx1250) — matrix operand order repaired and checked by reading, not compiled, not run on hardware
//
#include <hip/hip_runtime.h>

// ---------------- Problem constants ----------------
#define BATCH   64
#define TT      51          // T+1 timesteps
#define ROWS    (BATCH*TT)  // 3264
#define HID     512
#define G4      2048        // 4*HID
#define VOCAB   32000
#define NSLICE  16
#define SLICE_N 2000        // VOCAB / NSLICE
#define START_IDX 1
#define STOP_IDX  2

// ---------------- Types ----------------
typedef __bf16 bf16_t;
typedef __attribute__((ext_vector_type(16))) __bf16        v16bf;
typedef __attribute__((ext_vector_type(8)))  float         v8f;
typedef __attribute__((ext_vector_type(4)))  unsigned int  v4u;
typedef __attribute__((ext_vector_type(8)))  unsigned int  v8u;

__device__ __forceinline__ unsigned short f2bf(float f) {
    unsigned u = __builtin_bit_cast(unsigned, f);
    unsigned r = u + 0x7FFFu + ((u >> 16) & 1u);   // round-to-nearest-even
    return (unsigned short)(r >> 16);
}
__device__ __forceinline__ float bf2f(unsigned short s) {
    return __builtin_bit_cast(float, ((unsigned)s) << 16);
}

// ---------------- Async stage of one 16x512 bf16 A row-block into LDS ----------------
// 16*512*2B = 16KB = 1024 x b128 chunks, copied with GLOBAL_LOAD_ASYNC_TO_LDS_B128
// (CDNA5 ISA: per-lane LDS dest addr + global src addr, ASYNCcnt).
template <int NTHR>
__device__ __forceinline__ void stage16x512(unsigned short* sA, const unsigned short* g) {
    for (int c = threadIdx.x; c < 1024; c += NTHR) {
        unsigned lds_addr = (unsigned)(size_t)(sA + c * 8);      // low 32 bits = LDS byte addr
        const unsigned short* gp = g + c * 8;
        asm volatile("global_load_async_to_lds_b128 %0, %1, off"
                     :: "v"(lds_addr), "v"(gp) : "memory");
    }
    asm volatile("s_wait_asynccnt 0x0" ::: "memory");
}

// A-fragment from LDS-staged block: 16x32 bf16 (MxK). Lanes 0-15 -> M=lane,
// kb=0; lanes 16-31 -> M=lane-16, kb=16. Lane holds K = kb+[0..15].
__device__ __forceinline__ v16bf ldA_lds(const unsigned short* sA, int k0) {
    int lane = threadIdx.x & 31;
    int m    = lane & 15;
    int kb   = (lane >> 4) << 4;                    // 0 or 16
    const unsigned short* p = sA + m * HID + k0 + kb;
    v4u lo = *(const v4u*)p;                        // K = kb .. kb+7
    v4u hi = *(const v4u*)(p + 8);                  // K = kb+8 .. kb+15
    v8u r = { lo.x, lo.y, lo.z, lo.w, hi.x, hi.y, hi.z, hi.w };
    return __builtin_bit_cast(v16bf, r);
}

// B-fragment: 32x16 bf16 (KxN), B[k,n] = W[n,k] with W row-major [N,K].
// Same contiguous k order as ldA_lds: lanes 0-15 -> column n=lane, K=0..15; lanes 16-31 -> n=lane-16, K=16..31.
__device__ __forceinline__ v16bf ldB(const unsigned short* W, int ldw, int n0, int k0) {
    int lane = threadIdx.x & 31;
    int n    = lane & 15;
    int kb   = (lane >> 4) << 4;                    // 0 or 16
    const unsigned short* p = W + (size_t)(n0 + n) * ldw + k0 + kb;
    v4u lo = *(const v4u*)p;
    v4u hi = *(const v4u*)(p + 8);
    v8u r = { lo.x, lo.y, lo.z, lo.w, hi.x, hi.y, hi.z, hi.w };
    return __builtin_bit_cast(v16bf, r);
}

// ---------------- Small prep kernels ----------------
__global__ void k_cvt(unsigned short* __restrict__ dst, const float* __restrict__ src, int n) {
    for (int i = blockIdx.x * blockDim.x + threadIdx.x; i < n; i += gridDim.x * blockDim.x)
        dst[i] = f2bf(src[i]);
}

__global__ void k_bsum(float* __restrict__ bsum, const float* __restrict__ bih,
                       const float* __restrict__ bhh) {
    int i = blockIdx.x * blockDim.x + threadIdx.x;
    if (i < G4) bsum[i] = bih[i] + bhh[i];
}

// Xe[t*64+b, k] = bf16(emb[inputs[b,t], k]); inputs[b,0]=START, inputs[b,t]=labels[b,t-1]
__global__ void k_embed(unsigned short* __restrict__ Xe, const float* __restrict__ emb,
                        const int* __restrict__ labels) {
    int idx = blockIdx.x * blockDim.x + threadIdx.x;
    if (idx >= ROWS * HID) return;
    int r = idx >> 9, k = idx & (HID - 1);
    int t = r / BATCH, b = r % BATCH;
    int tok = (t == 0) ? START_IDX : labels[b * (TT - 1) + (t - 1)];
    Xe[idx] = f2bf(emb[(size_t)tok * HID + k]);
}

// c0 = x ; h0 = bf16(x)
__global__ void k_init(float* __restrict__ c, unsigned short* __restrict__ h0,
                       const float* __restrict__ x) {
    int i = blockIdx.x * blockDim.x + threadIdx.x;
    if (i < BATCH * HID) { float v = x[i]; c[i] = v; h0[i] = f2bf(v); }
}

// ---------------- GEMM 1: Xg[3264,2048] = Xe @ W_ih^T + (b_ih+b_hh) ----------------
__global__ __launch_bounds__(128) void k_gemm_xg(const unsigned short* __restrict__ Xe,
                                                 const unsigned short* __restrict__ Wih,
                                                 const float* __restrict__ bsum,
                                                 float* __restrict__ Xg) {
    __shared__ __align__(16) unsigned short sA[16 * HID];
    int wave = threadIdx.x >> 5;
    int lane = threadIdx.x & 31;
    int m0 = blockIdx.x * 16;                     // 204 row tiles
    int n0 = blockIdx.y * 64 + wave * 16;         // 32 col groups * 4 waves
    stage16x512<128>(sA, Xe + (size_t)m0 * HID);
    __syncthreads();
    float bn = bsum[n0 + (lane & 15)];
    v8f acc = { bn, bn, bn, bn, bn, bn, bn, bn };
#pragma unroll
    for (int k = 0; k < HID; k += 32) {
        v16bf a = ldA_lds(sA, k);
        v16bf b = ldB(Wih, HID, n0, k);
        acc = __builtin_amdgcn_wmma_f32_16x16x32_bf16(false, a, false, b, (short)0, acc, false, false);
    }
    int mrow = (lane >> 4) << 3;
    int n = n0 + (lane & 15);
#pragma unroll
    for (int v = 0; v < 8; ++v)
        Xg[(size_t)(m0 + mrow + v) * G4 + n] = acc[v];
}

// ---------------- LSTM step: gates = Xg[t] + h_prev @ W_hh^T ; cell update ----------------
// grid = 4 blocks (16 rows each), 256 threads (8 waves, 16 n-tiles per wave)
__global__ __launch_bounds__(256) void k_lstm_step(const unsigned short* __restrict__ hprev,
                                                   const unsigned short* __restrict__ Whh,
                                                   const float* __restrict__ Xg_t,
                                                   float* __restrict__ gates,
                                                   float* __restrict__ c,
                                                   unsigned short* __restrict__ hout) {
    __shared__ __align__(16) unsigned short sA[16 * HID];
    int wave = threadIdx.x >> 5;
    int lane = threadIdx.x & 31;
    int m0 = blockIdx.x * 16;
    int mrow = (lane >> 4) << 3;
    stage16x512<256>(sA, hprev + (size_t)m0 * HID);
    __syncthreads();
    for (int i = 0; i < 16; ++i) {
        int n0 = wave * 256 + i * 16;
        int n  = n0 + (lane & 15);
        v8f acc;
#pragma unroll
        for (int v = 0; v < 8; ++v)
            acc[v] = Xg_t[(size_t)(m0 + mrow + v) * G4 + n];
#pragma unroll
        for (int k = 0; k < HID; k += 32) {
            v16bf a = ldA_lds(sA, k);
            v16bf b = ldB(Whh, HID, n0, k);
            acc = __builtin_amdgcn_wmma_f32_16x16x32_bf16(false, a, false, b, (short)0, acc, false, false);
        }
#pragma unroll
        for (int v = 0; v < 8; ++v)
            gates[(size_t)(m0 + mrow + v) * G4 + n] = acc[v];
    }
    __syncthreads();
    // elementwise LSTM cell for this block's 16 rows
    for (int e = threadIdx.x; e < 16 * HID; e += 256) {
        int r = m0 + (e >> 9);
        int j = e & (HID - 1);
        const float* g = gates + (size_t)r * G4;
        float ig = g[j], fg = g[HID + j], gg = g[2 * HID + j], og = g[3 * HID + j];
        float iS = 1.f / (1.f + __expf(-ig));
        float fS = 1.f / (1.f + __expf(-fg));
        float gT = tanhf(gg);
        float oS = 1.f / (1.f + __expf(-og));
        float cn = fS * c[(size_t)r * HID + j] + iS * gT;
        c[(size_t)r * HID + j] = cn;
        hout[(size_t)r * HID + j] = f2bf(oS * tanhf(cn));
    }
}

// ---------------- Classifier GEMM fused with online logsumexp ----------------
// grid = (204 row-blocks, 16 vocab slices); 256 threads = 8 waves; logits never hit memory.
__global__ __launch_bounds__(256) void k_logits(const unsigned short* __restrict__ hs,
                                                const unsigned short* __restrict__ fcW,
                                                const float* __restrict__ fcb,
                                                float* __restrict__ part) {
    __shared__ __align__(16) unsigned short sA[16 * HID];
    __shared__ float s_m[8][16];
    __shared__ float s_s[8][16];
    int wave = threadIdx.x >> 5;
    int lane = threadIdx.x & 31;
    int m0 = blockIdx.x * 16;
    int slice = blockIdx.y;
    int nbase = slice * SLICE_N;
    stage16x512<256>(sA, hs + (size_t)m0 * HID);
    __syncthreads();
    float mx[8], sm[8];
#pragma unroll
    for (int v = 0; v < 8; ++v) { mx[v] = -3.0e38f; sm[v] = 0.f; }

    for (int tile = wave; tile < SLICE_N / 16; tile += 8) {
        int n0 = nbase + tile * 16;
        float bn = fcb[n0 + (lane & 15)];
        v8f acc = { bn, bn, bn, bn, bn, bn, bn, bn };
#pragma unroll
        for (int k = 0; k < HID; k += 32) {
            v16bf a = ldA_lds(sA, k);
            v16bf b = ldB(fcW, HID, n0, k);
            acc = __builtin_amdgcn_wmma_f32_16x16x32_bf16(false, a, false, b, (short)0, acc, false, false);
        }
        if (tile + 8 < SLICE_N / 16)   // hint next B tile into cache
            __builtin_prefetch(fcW + (size_t)(nbase + (tile + 8) * 16 + (lane & 15)) * HID, 0, 1);
#pragma unroll
        for (int v = 0; v < 8; ++v) {
            float xv = acc[v];
            float mn = fmaxf(mx[v], xv);
            sm[v] = sm[v] * __expf(mx[v] - mn) + __expf(xv - mn);
            mx[v] = mn;
        }
    }
    // combine across the 16 lanes of each half (rows live per-half)
#pragma unroll
    for (int d = 1; d < 16; d <<= 1) {
#pragma unroll
        for (int v = 0; v < 8; ++v) {
            float om = __shfl_xor(mx[v], d, 32);
            float os = __shfl_xor(sm[v], d, 32);
            float mn = fmaxf(mx[v], om);
            sm[v] = sm[v] * __expf(mx[v] - mn) + os * __expf(om - mn);
            mx[v] = mn;
        }
    }
    if (lane == 0) {
#pragma unroll
        for (int v = 0; v < 8; ++v) { s_m[wave][v] = mx[v]; s_s[wave][v] = sm[v]; }
    }
    if (lane == 16) {
#pragma unroll
        for (int v = 0; v < 8; ++v) { s_m[wave][8 + v] = mx[v]; s_s[wave][8 + v] = sm[v]; }
    }
    __syncthreads();
    if (threadIdx.x < 16) {
        int row = threadIdx.x;
        float M = s_m[0][row], S = s_s[0][row];
#pragma unroll
        for (int w = 1; w < 8; ++w) {
            float om = s_m[w][row], os = s_s[w][row];
            float mn = fmaxf(M, om);
            S = S * __expf(M - mn) + os * __expf(om - mn);
            M = mn;
        }
        size_t r = (size_t)(m0 + row);
        part[(r * NSLICE + slice) * 2 + 0] = M;
        part[(r * NSLICE + slice) * 2 + 1] = S;
    }
}

// ---------------- Per-row: combine slice partials, subtract target logit ----------------
__global__ __launch_bounds__(256) void k_reduce(const float* __restrict__ part,
                                                const unsigned short* __restrict__ hs,
                                                const unsigned short* __restrict__ fcW,
                                                const float* __restrict__ fcb,
                                                const int* __restrict__ labels,
                                                float* __restrict__ nll) {
    int wave = threadIdx.x >> 5;
    int lane = threadIdx.x & 31;
    int r = blockIdx.x * 8 + wave;
    if (r >= ROWS) return;
    float M = -3.0e38f, S = 0.f;
    if (lane < NSLICE) {
        M = part[((size_t)r * NSLICE + lane) * 2 + 0];
        S = part[((size_t)r * NSLICE + lane) * 2 + 1];
    }
#pragma unroll
    for (int d = 1; d < 16; d <<= 1) {
        float om = __shfl_xor(M, d, 32);
        float os = __shfl_xor(S, d, 32);
        float mn = fmaxf(M, om);
        S = S * __expf(M - mn) + os * __expf(om - mn);
        M = mn;
    }
    float lse = __shfl(M + __logf(S), 0, 32);
    int t = r / BATCH, b = r % BATCH;
    int tgt = (t < TT - 1) ? labels[b * (TT - 1) + t] : STOP_IDX;
    const unsigned short* hp = hs  + (size_t)r   * HID;
    const unsigned short* wp = fcW + (size_t)tgt * HID;
    float dot = 0.f;
#pragma unroll
    for (int i = 0; i < 16; ++i) {
        int k = lane * 16 + i;
        dot += bf2f(hp[k]) * bf2f(wp[k]);
    }
#pragma unroll
    for (int d = 1; d < 32; d <<= 1) dot += __shfl_xor(dot, d, 32);
    if (lane == 0) nll[r] = lse - (dot + fcb[tgt]);
}

// ---------------- Deterministic final sum ----------------
__global__ __launch_bounds__(256) void k_final(const float* __restrict__ nll, float* __restrict__ out) {
    __shared__ float red[256];
    float s = 0.f;
    for (int i = threadIdx.x; i < ROWS; i += 256) s += nll[i];
    red[threadIdx.x] = s;
    __syncthreads();
    for (int d = 128; d > 0; d >>= 1) {
        if (threadIdx.x < d) red[threadIdx.x] += red[threadIdx.x + d];
        __syncthreads();
    }
    if (threadIdx.x == 0) out[0] = red[0] / (float)BATCH;
}

// ---------------- Host launch ----------------
extern "C" void kernel_launch(void* const* d_in, const int* in_sizes, int n_in,
                              void* d_out, int out_size, void* d_ws, size_t ws_size,
                              hipStream_t stream) {
    const float* x      = (const float*)d_in[0];
    const int*   labels = (const int*)  d_in[1];
    const float* emb    = (const float*)d_in[2];
    const float* W_ih   = (const float*)d_in[3];
    const float* W_hh   = (const float*)d_in[4];
    const float* b_ih   = (const float*)d_in[5];
    const float* b_hh   = (const float*)d_in[6];
    const float* fc_W   = (const float*)d_in[7];
    const float* fc_b   = (const float*)d_in[8];
    float* out = (float*)d_out;

    char* ws = (char*)d_ws;
    size_t off = 0;
    auto take = [&](size_t bytes) -> char* {
        char* p = ws + off;
        off += (bytes + 255) & ~(size_t)255;
        return p;
    };
    unsigned short* Wih_b = (unsigned short*)take((size_t)G4 * HID * 2);
    unsigned short* Whh_b = (unsigned short*)take((size_t)G4 * HID * 2);
    unsigned short* fcW_b = (unsigned short*)take((size_t)VOCAB * HID * 2);
    unsigned short* Xe_b  = (unsigned short*)take((size_t)ROWS * HID * 2);
    float*          Xg    = (float*)take((size_t)ROWS * G4 * 4);
    unsigned short* hs    = (unsigned short*)take((size_t)ROWS * HID * 2);
    unsigned short* h0    = (unsigned short*)take((size_t)BATCH * HID * 2);
    float*          cst   = (float*)take((size_t)BATCH * HID * 4);
    float*          bsum  = (float*)take((size_t)G4 * 4);
    float*          gates = (float*)take((size_t)BATCH * G4 * 4);
    float*          part  = (float*)take((size_t)ROWS * NSLICE * 2 * 4);
    float*          nll   = (float*)take((size_t)ROWS * 4);

    // 1) precision conversion + bias fold + embedding gather + state init
    {
        int n1 = G4 * HID;
        k_cvt<<<(n1 + 255) / 256, 256, 0, stream>>>(Wih_b, W_ih, n1);
        k_cvt<<<(n1 + 255) / 256, 256, 0, stream>>>(Whh_b, W_hh, n1);
        int n2 = VOCAB * HID;
        k_cvt<<<(n2 + 255) / 256, 256, 0, stream>>>(fcW_b, fc_W, n2);
        k_bsum<<<(G4 + 255) / 256, 256, 0, stream>>>(bsum, b_ih, b_hh);
        int n3 = ROWS * HID;
        k_embed<<<(n3 + 255) / 256, 256, 0, stream>>>(Xe_b, emb, labels);
        k_init<<<(BATCH * HID + 255) / 256, 256, 0, stream>>>(cst, h0, x);
    }

    // 2) all-timestep input projection (WMMA bf16, A staged via async-to-LDS)
    k_gemm_xg<<<dim3(ROWS / 16, G4 / 64), 128, 0, stream>>>(Xe_b, Wih_b, bsum, Xg);

    // 3) sequential LSTM recurrence, one fused launch per timestep
    for (int t = 0; t < TT; ++t) {
        const unsigned short* hprev = (t == 0) ? h0 : (hs + (size_t)(t - 1) * BATCH * HID);
        k_lstm_step<<<BATCH / 16, 256, 0, stream>>>(
            hprev, Whh_b, Xg + (size_t)t * BATCH * G4, gates, cst,
            hs + (size_t)t * BATCH * HID);
    }

    // 4) classifier GEMM fused with online logsumexp (fc_W stays resident in 192MB L2)
    k_logits<<<dim3(ROWS / 16, NSLICE), 256, 0, stream>>>(hs, fcW_b, fc_b, part);

    // 5) per-row NLL, then deterministic scalar reduction
    k_reduce<<<(ROWS + 7) / 8, 256, 0, stream>>>(part, hs, fcW_b, fc_b, labels, nll);
    k_final<<<1, 256, 0, stream>>>(nll, out);
    (void)in_sizes; (void)n_in; (void)out_size; (void)ws_size;
}
